// S5SSM_7782480740554
// MI455X (gfx1250) — matrix operand order repaired and checked by reading, not compiled, not run on hardware
//
#include <hip/hip_runtime.h>

// ---------------- problem constants ----------------
#define LL    32768
#define HH    256
#define PP    256
#define CHUNK 128
#define NCH   (LL / CHUNK)   // 256 chunks

typedef __attribute__((ext_vector_type(16))) __bf16 v16bf;
typedef __attribute__((ext_vector_type(8)))  __bf16 v8bf;
typedef __attribute__((ext_vector_type(4)))  __bf16 v4bf;
typedef __attribute__((ext_vector_type(8)))  float  v8f;

// ---------------------------------------------------------------------------
// WMMA fragment loaders (wave32, v_wmma_f32_16x16x32_bf16 layouts)
// A (16x32, MxK): lanes 0-15 hold M=lane, K={k0..k0+7, k0+16..k0+23};
//                 lanes 16-31 hold M=lane-16, K={k0+8..k0+15, k0+24..k0+31}.
// B (32x16, KxN): lanes 0-15 hold N=lane, K={k0..k0+7, k0+16..k0+23};
//                 lanes 16-31 hold N=lane-16, K={k0+8..k0+15, k0+24..k0+31}.
// ---------------------------------------------------------------------------
__device__ __forceinline__ v16bf load_a_frag(const __bf16* __restrict__ base,
                                             int stride, int row0, int k0, int lane) {
    const int half = lane >> 4;
    const int row  = row0 + (lane & 15);
    const __bf16* p = base + (size_t)row * stride + k0 + half * 8;
    union { v16bf v; v8bf h[2]; } u;
    u.h[0] = *(const v8bf*)(p);
    u.h[1] = *(const v8bf*)(p + 16);
    return u.v;
}

// B fragment out of an LDS tile laid out as [64 cols][256 K] bf16 (row-major K).
__device__ __forceinline__ v16bf lds_b_frag(const __bf16* lds,
                                            int col0, int k0, int lane) {
    const int half = lane >> 4;
    const int col  = col0 + (lane & 15);
    const __bf16* p = lds + col * 256 + k0 + half * 8;
    union { v16bf v; v8bf h[2]; } u;
    u.h[0] = *(const v8bf*)(p);
    u.h[1] = *(const v8bf*)(p + 16);
    return u.v;
}

// ---------------------------------------------------------------------------
// Cooperative async DMA of a (64 rows x 256 bf16) global tile into LDS.
// 64 * 512B = 2048 sixteen-byte chunks, 8 per thread (256 threads).
// Uses CDNA5 GLOBAL_LOAD_ASYNC_TO_LDS_B128 (ASYNCcnt-tracked).
// ---------------------------------------------------------------------------
__device__ __forceinline__ void async_tile_to_lds(unsigned lds_base,
                                                  const __bf16* gbase,
                                                  int row0, int stride_elems,
                                                  int tid) {
#pragma unroll
    for (int q = 0; q < 8; ++q) {
        const int chunk = tid * 8 + q;           // 0..2047
        const int row   = chunk >> 5;            // chunk / 32
        const int c16   = chunk & 31;            // 16B chunk within row
        const __bf16* g = gbase + (size_t)(row0 + row) * stride_elems + c16 * 8;
        const unsigned lds = lds_base + row * 512 + c16 * 16;
        asm volatile("global_load_async_to_lds_b128 %0, %1, off"
                     :: "v"(lds), "v"(g) : "memory");
    }
}

__device__ __forceinline__ void wait_async_and_sync() {
    asm volatile("s_wait_asynccnt 0x0" ::: "memory");
    __syncthreads();
}

// ---------------------------------------------------------------------------
// Kernel 1: discretization + pack operands to bf16 planes.
// ---------------------------------------------------------------------------
__global__ __launch_bounds__(256) void k_precompute(
    const float* __restrict__ Lam, const float* __restrict__ Bri,
    const float* __restrict__ Cri, const float* __restrict__ logstep,
    __bf16* __restrict__ bbr, __bf16* __restrict__ bbi,
    __bf16* __restrict__ cr,  __bf16* __restrict__ cmi,
    float* __restrict__ lam_re, float* __restrict__ lam_im,
    float* __restrict__ lamT_re, float* __restrict__ lamT_im)
{
    const int t = blockIdx.x * 256 + threadIdx.x;   // 0 .. P*H-1
    const int p = t >> 8;                           // H == 256
    const float lr = Lam[2 * p], li = Lam[2 * p + 1];
    const float dt = __expf(logstep[p]);
    const float er = __expf(lr * dt);
    const float lbr = er * __cosf(li * dt);         // Lambda_bar
    const float lbi = er * __sinf(li * dt);
    // (Lambda_bar - 1) / Lambda
    const float inv = 1.0f / (lr * lr + li * li);
    const float nr = lbr - 1.0f, ni = lbi;
    const float sr = (nr * lr + ni * li) * inv;
    const float si = (ni * lr - nr * li) * inv;
    const float br = Bri[2 * t], bi = Bri[2 * t + 1];   // t == p*H+h
    bbr[t] = (__bf16)(sr * br - si * bi);
    bbi[t] = (__bf16)(sr * bi + si * br);
    // C planes: same flat index, t == h*P+p
    cr[t]  = (__bf16)(Cri[2 * t]);
    cmi[t] = (__bf16)(-Cri[2 * t + 1]);

    if (t < PP) {
        const float lr2 = Lam[2 * t], li2 = Lam[2 * t + 1];
        const float dt2 = __expf(logstep[t]);
        const float e1  = __expf(lr2 * dt2);
        lam_re[t] = e1 * __cosf(li2 * dt2);
        lam_im[t] = e1 * __sinf(li2 * dt2);
        const float aT = (float)CHUNK * lr2 * dt2;
        const float bT = (float)CHUNK * li2 * dt2;
        const float eT = __expf(aT);
        lamT_re[t] = eT * __cosf(bT);
        lamT_im[t] = eT * __sinf(bT);
    }
}

// ---------------------------------------------------------------------------
// Kernel 2: signal fp32 -> bf16 plane (L,H), 4 elems/thread
// ---------------------------------------------------------------------------
__global__ __launch_bounds__(256) void k_sig2bf(const float4* __restrict__ in,
                                                v4bf* __restrict__ out)
{
    const int t = blockIdx.x * 256 + threadIdx.x;
    const float4 v = in[t];
    v4bf o;
    o[0] = (__bf16)v.x; o[1] = (__bf16)v.y;
    o[2] = (__bf16)v.z; o[3] = (__bf16)v.w;
    out[t] = o;
}

// ---------------------------------------------------------------------------
// Kernel 3: Bu = sig(L,H) @ B_bar(P,H)^T   (re & im planes, fp32 out)
// B tiles (64 x 256 x 2 planes = 64KB) async-DMA'd to LDS once per block.
// Each wave owns TWO M-tiles (M=32): every LDS B-fragment feeds 2 WMMAs,
// halving the ds-load-per-wmma ratio so LDS latency hides behind matrix work.
// ---------------------------------------------------------------------------
__global__ __launch_bounds__(256) void k_gemm_bu(
    const __bf16* __restrict__ sigb,  // (L,H)
    const __bf16* __restrict__ bbr,   // (P,H)
    const __bf16* __restrict__ bbi,   // (P,H)
    float* __restrict__ bur,          // (L,P)
    float* __restrict__ bui)          // (L,P)
{
    __shared__ __bf16 sBr[64 * 256];  // 32KB
    __shared__ __bf16 sBi[64 * 256];  // 32KB

    const int tid = threadIdx.x;
    const int n0  = blockIdx.y * 64;

    async_tile_to_lds((unsigned)(uintptr_t)(void*)sBr, bbr, n0, HH, tid);
    async_tile_to_lds((unsigned)(uintptr_t)(void*)sBi, bbi, n0, HH, tid);
    wait_async_and_sync();

    const int lane = tid & 31;
    const int wave = tid >> 5;
    const int m0 = blockIdx.x * 256 + wave * 32;   // two 16-row tiles per wave

    v8f accR[2][4], accI[2][4];
    const v8f vzero = {};
#pragma unroll
    for (int m = 0; m < 2; ++m)
#pragma unroll
        for (int j = 0; j < 4; ++j) { accR[m][j] = vzero; accI[m][j] = vzero; }

    for (int kk = 0; kk < HH; kk += 32) {
        __builtin_prefetch(sigb + (size_t)(m0 + (lane & 15)) * HH + kk + 64, 0, 1);
        const v16bf a0 = load_a_frag(sigb, HH, m0,      kk, lane);
        const v16bf a1 = load_a_frag(sigb, HH, m0 + 16, kk, lane);
#pragma unroll
        for (int j = 0; j < 4; ++j) {
            const v16bf br = lds_b_frag(sBr, j * 16, kk, lane);
            accR[0][j] = __builtin_amdgcn_wmma_f32_16x16x32_bf16(
                false, a0, false, br, (short)0, accR[0][j], false, false);
            accR[1][j] = __builtin_amdgcn_wmma_f32_16x16x32_bf16(
                false, a1, false, br, (short)0, accR[1][j], false, false);
            const v16bf bi = lds_b_frag(sBi, j * 16, kk, lane);
            accI[0][j] = __builtin_amdgcn_wmma_f32_16x16x32_bf16(
                false, a0, false, bi, (short)0, accI[0][j], false, false);
            accI[1][j] = __builtin_amdgcn_wmma_f32_16x16x32_bf16(
                false, a1, false, bi, (short)0, accI[1][j], false, false);
        }
    }

    const int half = lane >> 4, nc = lane & 15;
#pragma unroll
    for (int m = 0; m < 2; ++m)
#pragma unroll
        for (int j = 0; j < 4; ++j)
#pragma unroll
            for (int r = 0; r < 8; ++r) {
                const int row = m0 + m * 16 + half * 8 + r;
                const size_t idx = (size_t)row * PP + n0 + j * 16 + nc;
                bur[idx] = accR[m][j][r];
                bui[idx] = accI[m][j][r];
            }
}

// ---------------------------------------------------------------------------
// Kernel 4: per-chunk local scan (in place), constant coefficient Lambda_bar.
// ---------------------------------------------------------------------------
__global__ __launch_bounds__(256) void k_scan_local(
    float* __restrict__ bur, float* __restrict__ bui,
    const float* __restrict__ lam_re, const float* __restrict__ lam_im,
    float* __restrict__ car, float* __restrict__ cai)
{
    const int p = threadIdx.x, c = blockIdx.x;
    const float ar = lam_re[p], ai = lam_im[p];
    float xr = 0.f, xi = 0.f;
    const size_t base = (size_t)c * CHUNK * PP + p;
    for (int i = 0; i < CHUNK; ++i) {
        const size_t idx = base + (size_t)i * PP;
        const float br = bur[idx], bi = bui[idx];
        const float nxr = ar * xr - ai * xi + br;
        const float nxi = ar * xi + ai * xr + bi;
        xr = nxr; xi = nxi;
        bur[idx] = xr; bui[idx] = xi;
    }
    car[c * PP + p] = xr;
    cai[c * PP + p] = xi;
}

// ---------------------------------------------------------------------------
// Kernel 5: scan the 256 chunk carries (multiplier = Lambda_bar^CHUNK).
// Writes EXCLUSIVE prefix (state entering each chunk). Single block.
// ---------------------------------------------------------------------------
__global__ __launch_bounds__(256) void k_scan_carries(
    const float* __restrict__ car, const float* __restrict__ cai,
    const float* __restrict__ lamT_re, const float* __restrict__ lamT_im,
    float* __restrict__ pre_r, float* __restrict__ pre_i)
{
    const int p = threadIdx.x;
    const float tr = lamT_re[p], ti = lamT_im[p];
    float sr = 0.f, si = 0.f;
    for (int c = 0; c < NCH; ++c) {
        pre_r[c * PP + p] = sr;
        pre_i[c * PP + p] = si;
        const float crr = car[c * PP + p], cii = cai[c * PP + p];
        const float nr = tr * sr - ti * si + crr;
        const float ni = tr * si + ti * sr + cii;
        sr = nr; si = ni;
    }
}

// ---------------------------------------------------------------------------
// Kernel 6: x = x_local + Lambda_bar^{i+1} * prefix ; convert to bf16 planes.
// ---------------------------------------------------------------------------
__global__ __launch_bounds__(256) void k_fixup(
    const float* __restrict__ bur, const float* __restrict__ bui,
    const float* __restrict__ pre_r, const float* __restrict__ pre_i,
    const float* __restrict__ lam_re, const float* __restrict__ lam_im,
    __bf16* __restrict__ xrb, __bf16* __restrict__ xib)
{
    const int p = threadIdx.x, c = blockIdx.x;
    const float ar = lam_re[p], ai = lam_im[p];
    const float pr = pre_r[c * PP + p], pi = pre_i[c * PP + p];
    float cr = ar, ci = ai;                  // Lambda_bar^1
    const size_t base = (size_t)c * CHUNK * PP + p;
    for (int i = 0; i < CHUNK; ++i) {
        const size_t idx = base + (size_t)i * PP;
        const float xr = bur[idx] + cr * pr - ci * pi;
        const float xi = bui[idx] + cr * pi + ci * pr;
        xrb[idx] = (__bf16)xr;
        xib[idx] = (__bf16)xi;
        const float ncr = cr * ar - ci * ai;
        const float nci = cr * ai + ci * ar;
        cr = ncr; ci = nci;
    }
}

// ---------------------------------------------------------------------------
// Kernel 7: y = xr(L,P)@Cr(H,P)^T + xi@(-Ci)^T + D*u   (fp32 out)
// C tiles async-DMA'd to LDS; two M-tiles per wave, same as k_gemm_bu.
// ---------------------------------------------------------------------------
__global__ __launch_bounds__(256) void k_gemm_y(
    const __bf16* __restrict__ xrb,   // (L,P)
    const __bf16* __restrict__ xib,   // (L,P)
    const __bf16* __restrict__ crb,   // (H,P)
    const __bf16* __restrict__ cmib,  // (H,P)
    const float* __restrict__ sig,    // (L,H)
    const float* __restrict__ Dv,     // (H,)
    float* __restrict__ y)            // (L,H)
{
    __shared__ __bf16 sCr[64 * 256];  // 32KB
    __shared__ __bf16 sCi[64 * 256];  // 32KB

    const int tid = threadIdx.x;
    const int n0  = blockIdx.y * 64;

    async_tile_to_lds((unsigned)(uintptr_t)(void*)sCr, crb,  n0, PP, tid);
    async_tile_to_lds((unsigned)(uintptr_t)(void*)sCi, cmib, n0, PP, tid);
    wait_async_and_sync();

    const int lane = tid & 31;
    const int wave = tid >> 5;
    const int m0 = blockIdx.x * 256 + wave * 32;   // two 16-row tiles per wave

    v8f acc[2][4];
    const v8f vzero = {};
#pragma unroll
    for (int m = 0; m < 2; ++m)
#pragma unroll
        for (int j = 0; j < 4; ++j) acc[m][j] = vzero;

    for (int kk = 0; kk < PP; kk += 32) {
        __builtin_prefetch(xrb + (size_t)(m0 + (lane & 15)) * PP + kk + 64, 0, 1);
        const v16bf ar0 = load_a_frag(xrb, PP, m0,      kk, lane);
        const v16bf ar1 = load_a_frag(xrb, PP, m0 + 16, kk, lane);
        const v16bf ai0 = load_a_frag(xib, PP, m0,      kk, lane);
        const v16bf ai1 = load_a_frag(xib, PP, m0 + 16, kk, lane);
#pragma unroll
        for (int j = 0; j < 4; ++j) {
            const v16bf b1 = lds_b_frag(sCr, j * 16, kk, lane);
            acc[0][j] = __builtin_amdgcn_wmma_f32_16x16x32_bf16(
                false, ar0, false, b1, (short)0, acc[0][j], false, false);
            acc[1][j] = __builtin_amdgcn_wmma_f32_16x16x32_bf16(
                false, ar1, false, b1, (short)0, acc[1][j], false, false);
            const v16bf b2 = lds_b_frag(sCi, j * 16, kk, lane);
            acc[0][j] = __builtin_amdgcn_wmma_f32_16x16x32_bf16(
                false, ai0, false, b2, (short)0, acc[0][j], false, false);
            acc[1][j] = __builtin_amdgcn_wmma_f32_16x16x32_bf16(
                false, ai1, false, b2, (short)0, acc[1][j], false, false);
        }
    }

    const int half = lane >> 4, nc = lane & 15;
#pragma unroll
    for (int m = 0; m < 2; ++m)
#pragma unroll
        for (int j = 0; j < 4; ++j)
#pragma unroll
            for (int r = 0; r < 8; ++r) {
                const int row = m0 + m * 16 + half * 8 + r;
                const int col = n0 + j * 16 + nc;
                const size_t idx = (size_t)row * HH + col;
                y[idx] = acc[m][j][r] + Dv[col] * sig[idx];
            }
}

// ---------------------------------------------------------------------------
extern "C" void kernel_launch(void* const* d_in, const int* in_sizes, int n_in,
                              void* d_out, int out_size, void* d_ws, size_t ws_size,
                              hipStream_t stream)
{
    (void)in_sizes; (void)n_in; (void)out_size; (void)ws_size;

    const float* sig  = (const float*)d_in[0];   // (L,H)
    const float* Lam  = (const float*)d_in[1];   // (P,2)
    const float* Bri  = (const float*)d_in[2];   // (P,H,2)
    const float* Cri  = (const float*)d_in[3];   // (H,P,2)
    const float* Dv   = (const float*)d_in[4];   // (H,)
    const float* lstp = (const float*)d_in[5];   // (P,)
    float* y = (float*)d_out;

    char* w = (char*)d_ws;
    size_t o = 0;
    auto take = [&](size_t bytes) -> char* {
        char* p = w + o;
        o += (bytes + 255) & ~(size_t)255;
        return p;
    };

    __bf16* bbr  = (__bf16*)take((size_t)PP * HH * 2);       // B_bar re (bf16)
    __bf16* bbi  = (__bf16*)take((size_t)PP * HH * 2);       // B_bar im
    __bf16* crb  = (__bf16*)take((size_t)HH * PP * 2);       // C re
    __bf16* cmib = (__bf16*)take((size_t)HH * PP * 2);       // -C im
    float* lam_re  = (float*)take(PP * 4);
    float* lam_im  = (float*)take(PP * 4);
    float* lamT_re = (float*)take(PP * 4);
    float* lamT_im = (float*)take(PP * 4);
    __bf16* sigb = (__bf16*)take((size_t)LL * HH * 2);       // signal bf16
    float* bur   = (float*)take((size_t)LL * PP * 4);        // Bu / x_local re
    float* bui   = (float*)take((size_t)LL * PP * 4);        // Bu / x_local im
    float* car   = (float*)take((size_t)NCH * PP * 4);
    float* cai   = (float*)take((size_t)NCH * PP * 4);
    float* pre_r = (float*)take((size_t)NCH * PP * 4);
    float* pre_i = (float*)take((size_t)NCH * PP * 4);
    __bf16* xrb  = (__bf16*)take((size_t)LL * PP * 2);       // x re bf16
    __bf16* xib  = (__bf16*)take((size_t)LL * PP * 2);       // x im bf16

    k_precompute<<<(PP * HH) / 256, 256, 0, stream>>>(
        Lam, Bri, Cri, lstp, bbr, bbi, crb, cmib,
        lam_re, lam_im, lamT_re, lamT_im);

    k_sig2bf<<<(LL * HH / 4) / 256, 256, 0, stream>>>(
        (const float4*)sig, (v4bf*)sigb);

    k_gemm_bu<<<dim3(LL / 256, PP / 64), 256, 0, stream>>>(
        sigb, bbr, bbi, bur, bui);

    k_scan_local<<<NCH, PP, 0, stream>>>(bur, bui, lam_re, lam_im, car, cai);

    k_scan_carries<<<1, PP, 0, stream>>>(car, cai, lamT_re, lamT_im, pre_r, pre_i);

    k_fixup<<<NCH, PP, 0, stream>>>(bur, bui, pre_r, pre_i,
                                    lam_re, lam_im, xrb, xib);

    k_gemm_y<<<dim3(LL / 256, HH / 64), 256, 0, stream>>>(
        xrb, xib, crb, cmib, sig, Dv, y);
}
